// CustomBLIP_44650480009900
// MI455X (gfx1250) — matrix operand order repaired and checked by reading, not compiled, not run on hardware
//
#include <hip/hip_runtime.h>
#include <stdint.h>

// ---------------------------------------------------------------------------
// CustomBLIP fused bilinear head for MI455X (gfx1250, wave32, WMMA).
// Roofline: Wp stream (512MB fp32, read exactly once) => ~22us @ 23.3TB/s.
// bf16 WMMA 16x16x32 compute ~14us on the matrix pipe; all VALU/LDS helper
// work is kept under the HBM floor (4 pkmul + 1 ds_load_b128 per WMMA) and
// co-executes with the XDL pipe. Wp loads are software-pipelined one group
// ahead in registers to hide HBM latency.
// ---------------------------------------------------------------------------

#define SEGS      16          // r-segments (each = 32 'i' rows of 512 j's)
#define I_PER_SEG 32
#define IG        4           // i-rows staged per pipeline group
#define JC        64          // j-chunk staged per outer step
#define KW        32          // k-tile width per block (two 16-wide n-tiles)
#define TXT_STR   36          // LDS row stride in dwords (bf16 pairs), padded
#define WP_STR    36

typedef __attribute__((ext_vector_type(16))) __bf16   v16bf;
typedef __attribute__((ext_vector_type(8)))  float    v8f;
typedef __attribute__((ext_vector_type(4)))  float    v4f;
typedef __attribute__((ext_vector_type(4)))  uint32_t v4u;

union ABFrag { uint32_t u[8]; v4u q[2]; v16bf bf; };

// One-instruction f32x2 -> packed bf16x2 (RNE).
__device__ __forceinline__ uint32_t pack2(float lo, float hi) {
  uint32_t d;
  asm("v_cvt_pk_bf16_f32 %0, %1, %2" : "=v"(d) : "v"(lo), "v"(hi));
  return d;
}
__device__ __forceinline__ uint32_t pkmul(uint32_t a, uint32_t b) {
  uint32_t d;
  asm("v_pk_mul_bf16 %0, %1, %2" : "=v"(d) : "v"(a), "v"(b));
  return d;
}

// ---------------------------------------------------------------------------
// Kernel 1: y = X @ W^T + bias, L2-normalize rows.
// which==0 -> image path, stores TRANSPOSED imgT[k][b] (coalesced i-column
// loads in the bilinear kernel). which==1 -> text path, row-major txtN[b][k].
// ---------------------------------------------------------------------------
__global__ __launch_bounds__(256) void map_norm_kernel(
    const float* __restrict__ img_e, const float* __restrict__ txt_e,
    const float* __restrict__ Wi, const float* __restrict__ bi,
    const float* __restrict__ Wt, const float* __restrict__ bt,
    float* __restrict__ imgT, float* __restrict__ txtN)
{
  const int which = blockIdx.y;
  const float* X    = which ? txt_e : img_e;
  const float* W    = which ? Wt    : Wi;
  const float* bias = which ? bt    : bi;
  const int b0 = blockIdx.x * 16;
  const int t  = threadIdx.x;

  __shared__ float xt[16 * 512];    // 32 KB: 16 input rows
  __shared__ float red[16 * 256];   // 16 KB: per-row sum-of-squares partials

  #pragma unroll
  for (int q = 0; q < 8; ++q) {
    int idx = q * 1024 + t * 4;
    *(v4f*)(xt + idx) = *(const v4f*)(X + (size_t)b0 * 512 + idx);
  }
  __syncthreads();

  float a0[16], a1[16];
  #pragma unroll
  for (int b = 0; b < 16; ++b) { a0[b] = 0.f; a1[b] = 0.f; }

  const float* w0p = W + (size_t)t * 512;          // output col k = t
  const float* w1p = W + (size_t)(t + 256) * 512;  // output col k = t+256
  for (int p = 0; p < 512; p += 4) {
    v4f wa = *(const v4f*)(w0p + p);
    v4f wb = *(const v4f*)(w1p + p);
    #pragma unroll
    for (int b = 0; b < 16; ++b) {
      v4f xv = *(const v4f*)(xt + b * 512 + p);    // LDS broadcast
      a0[b] += xv.x * wa.x + xv.y * wa.y + xv.z * wa.z + xv.w * wa.w;
      a1[b] += xv.x * wb.x + xv.y * wb.y + xv.z * wb.z + xv.w * wb.w;
    }
  }
  const float bia0 = bias[t], bia1 = bias[t + 256];
  #pragma unroll
  for (int b = 0; b < 16; ++b) {
    a0[b] += bia0; a1[b] += bia1;
    red[b * 256 + t] = a0[b] * a0[b] + a1[b] * a1[b];
  }
  __syncthreads();
  for (int s = 128; s >= 1; s >>= 1) {       // deterministic tree reduce
    if (t < s) {
      #pragma unroll
      for (int b = 0; b < 16; ++b) red[b * 256 + t] += red[b * 256 + t + s];
    }
    __syncthreads();
  }
  #pragma unroll
  for (int b = 0; b < 16; ++b) {
    float sc = 1.f / fmaxf(sqrtf(red[b * 256]), 1e-12f);
    float y0 = a0[b] * sc, y1 = a1[b] * sc;
    if (which == 0) {
      imgT[(size_t)t * 256 + b0 + b]         = y0;
      imgT[(size_t)(t + 256) * 256 + b0 + b] = y1;
    } else {
      txtN[(size_t)(b0 + b) * 512 + t]       = y0;
      txtN[(size_t)(b0 + b) * 512 + t + 256] = y1;
    }
  }
}

// ---------------------------------------------------------------------------
// Kernel 2: pfeats[seg,b,k] = sum_{r in seg} img[b,i]*txt[b,j]*Wp[k,r]
// Block = (32-wide k-tile) x (r-segment), all 256 b rows -> Wp read once.
// 8 waves x 2 b-subtiles x 16 rows = 256 b; per wave 4 accumulators (2 sub x
// 2 k-halves). txt fragments hoisted to registers per j-chunk; Wp groups of
// IG=4 i-rows staged through LDS with one-group-ahead register prefetch.
// ---------------------------------------------------------------------------
__global__ __launch_bounds__(256) void bilinear_kernel(
    const float* __restrict__ Wp, const float* __restrict__ imgT,
    const float* __restrict__ txtN, float* __restrict__ pfeats)
{
  __shared__ uint32_t lds_txt[256 * TXT_STR];      // bf16 pairs txt[b][j-chunk]
  __shared__ uint32_t lds_wp [IG * KW * WP_STR];   // bf16 pairs Wp[(q,kk)][j]

  const int t    = threadIdx.x;
  const int lane = t & 31;
  const int wid  = t >> 5;           // wave 0..7
  const int m    = lane & 15;        // A row / B col / D col within tile
  const int h    = lane >> 4;        // lane half

  const int kt  = blockIdx.x & 15;
  const int seg = blockIdx.x >> 4;
  const int k0  = kt * KW;
  const int i_base = seg * I_PER_SEG;
  const int b0  = wid * 32;          // this wave owns b rows [b0, b0+32)

  // Wp staging role of this thread: row kk (0..31) of k-tile, 8 floats at c8.
  const int kk = t >> 3;
  const int c8 = (t & 7) * 8;
  const float* wp_base = Wp + (size_t)(k0 + kk) * (512 * 512) + c8;

  v8f acc00 = {}, acc01 = {}, acc10 = {}, acc11 = {}; // [sub][k-half]

  for (int j0 = 0; j0 < 512; j0 += JC) {
    __syncthreads();                 // all txt-frag hoists of prev chunk done
    // --- prefetch pipeline group 0 of this j-chunk (overlaps txt staging) ---
    v4f  wv[IG][2];
    float iv[IG];
    #pragma unroll
    for (int q = 0; q < IG; ++q) {
      const size_t ro = (size_t)(i_base + q) * 512 + j0;
      wv[q][0] = __builtin_nontemporal_load((const v4f*)(wp_base + ro));
      wv[q][1] = __builtin_nontemporal_load((const v4f*)(wp_base + ro + 4));
      iv[q]    = imgT[(size_t)(i_base + q) * 256 + t];
    }
    // --- stage txt tile: thread t = row b, JC floats -> bf16 pairs ---
    {
      const float* src = txtN + (size_t)t * 512 + j0;
      #pragma unroll
      for (int qq = 0; qq < JC / 4; ++qq) {
        v4f v = *(const v4f*)(src + 4 * qq);
        lds_txt[t * TXT_STR + 2 * qq]     = pack2(v.x, v.y);
        lds_txt[t * TXT_STR + 2 * qq + 1] = pack2(v.z, v.w);
      }
    }
    __syncthreads();
    // --- hoist txt fragments (invariant across all i of this j-chunk) ---
    v4u ta[2][2][2];                 // [sub][js][run]
    #pragma unroll
    for (int sub = 0; sub < 2; ++sub) {
      const uint32_t* trow = lds_txt + (b0 + sub * 16 + m) * TXT_STR;
      #pragma unroll
      for (int js = 0; js < 2; ++js) {
        ta[sub][js][0] = *(const v4u*)(trow + js * 16 + 4 * h);
        ta[sub][js][1] = *(const v4u*)(trow + js * 16 + 8 + 4 * h);
      }
    }

    for (int ig = 0; ig < I_PER_SEG / IG; ++ig) {
      __syncthreads();               // prev group's lds_wp readers done
      // --- store staged group to LDS; make replicated img bf16 pairs ---
      uint32_t ipair[IG];
      #pragma unroll
      for (int q = 0; q < IG; ++q) {
        v4u pk = { pack2(wv[q][0].x, wv[q][0].y), pack2(wv[q][0].z, wv[q][0].w),
                   pack2(wv[q][1].x, wv[q][1].y), pack2(wv[q][1].z, wv[q][1].w) };
        *(v4u*)(lds_wp + (q * KW + kk) * WP_STR + (c8 >> 1)) = pk;
        ipair[q] = pack2(iv[q], iv[q]);
      }
      __syncthreads();
      // --- prefetch next group into registers (hides HBM latency) ---
      if (ig + 1 < I_PER_SEG / IG) {
        const int inext = i_base + (ig + 1) * IG;
        #pragma unroll
        for (int q = 0; q < IG; ++q) {
          const size_t ro = (size_t)(inext + q) * 512 + j0;
          wv[q][0] = __builtin_nontemporal_load((const v4f*)(wp_base + ro));
          wv[q][1] = __builtin_nontemporal_load((const v4f*)(wp_base + ro + 4));
          iv[q]    = imgT[(size_t)(inext + q) * 256 + t];
        }
      }
      // --- compute group: IG i-rows x 2 K-steps x (2 sub x 2 k-half) WMMA ---
      #pragma unroll
      for (int q = 0; q < IG; ++q) {
        // wave's own lanes hold img[b0..b0+31]; broadcast via lane permute
        const uint32_t imgp0 = (uint32_t)__shfl((int)ipair[q], m, 32);
        const uint32_t imgp1 = (uint32_t)__shfl((int)ipair[q], 16 + m, 32);
        #pragma unroll
        for (int js = 0; js < 2; ++js) {
          ABFrag af0, af1;
          #pragma unroll
          for (int r = 0; r < 2; ++r) {
            v4u s0 = ta[0][js][r], s1 = ta[1][js][r];
            af0.u[4*r+0] = pkmul(s0.x, imgp0); af0.u[4*r+1] = pkmul(s0.y, imgp0);
            af0.u[4*r+2] = pkmul(s0.z, imgp0); af0.u[4*r+3] = pkmul(s0.w, imgp0);
            af1.u[4*r+0] = pkmul(s1.x, imgp1); af1.u[4*r+1] = pkmul(s1.y, imgp1);
            af1.u[4*r+2] = pkmul(s1.z, imgp1); af1.u[4*r+3] = pkmul(s1.w, imgp1);
          }
          // k-half 0
          {
            const uint32_t* wrow = lds_wp + (q * KW + m) * WP_STR;
            ABFrag bfr;
            bfr.q[0] = *(const v4u*)(wrow + js * 16 + 4 * h);
            bfr.q[1] = *(const v4u*)(wrow + js * 16 + 8 + 4 * h);
            acc00 = __builtin_amdgcn_wmma_f32_16x16x32_bf16(
                false, af0.bf, false, bfr.bf, (short)0, acc00, false, false);
            acc10 = __builtin_amdgcn_wmma_f32_16x16x32_bf16(
                false, af1.bf, false, bfr.bf, (short)0, acc10, false, false);
          }
          // k-half 1
          {
            const uint32_t* wrow = lds_wp + (q * KW + 16 + m) * WP_STR;
            ABFrag bfr;
            bfr.q[0] = *(const v4u*)(wrow + js * 16 + 4 * h);
            bfr.q[1] = *(const v4u*)(wrow + js * 16 + 8 + 4 * h);
            acc01 = __builtin_amdgcn_wmma_f32_16x16x32_bf16(
                false, af0.bf, false, bfr.bf, (short)0, acc01, false, false);
            acc11 = __builtin_amdgcn_wmma_f32_16x16x32_bf16(
                false, af1.bf, false, bfr.bf, (short)0, acc11, false, false);
          }
        }
      }
    }
  }
  // Epilogue: D layout -> row M = v + 8h, col N = m. Unique writes, no atomics.
  float* dst = pfeats + (size_t)seg * (256 * 512);
  #pragma unroll
  for (int v = 0; v < 8; ++v) {
    const int r0 = b0 + v + 8 * h;
    const int r1 = b0 + 16 + v + 8 * h;
    dst[(size_t)r0 * 512 + k0 + m]      = acc00[v];
    dst[(size_t)r0 * 512 + k0 + 16 + m] = acc01[v];
    dst[(size_t)r1 * 512 + k0 + m]      = acc10[v];
    dst[(size_t)r1 * 512 + k0 + 16 + m] = acc11[v];
  }
}

// ---------------------------------------------------------------------------
// Kernel 3: feats = sum_seg + bp, ReLU, logits = feats @ Wc^T + bc, sigmoid.
// ---------------------------------------------------------------------------
__global__ __launch_bounds__(512) void head_kernel(
    const float* __restrict__ pfeats, const float* __restrict__ bp,
    const float* __restrict__ Wc, const float* __restrict__ bc,
    float* __restrict__ out)
{
  __shared__ float red[512];
  const int b = blockIdx.x;
  const int k = threadIdx.x;
  float f = 0.f;
  #pragma unroll
  for (int s = 0; s < SEGS; ++s)
    f += pfeats[((size_t)s * 256 + b) * 512 + k];
  f += bp[k];
  f = fmaxf(f, 0.f);
  red[k] = f * Wc[k];
  __syncthreads();
  for (int s = 256; s >= 1; s >>= 1) {
    if (k < s) red[k] += red[k + s];
    __syncthreads();
  }
  if (k == 0) out[b] = 1.f / (1.f + expf(-(red[0] + bc[0])));
}

// ---------------------------------------------------------------------------
extern "C" void kernel_launch(void* const* d_in, const int* in_sizes, int n_in,
                              void* d_out, int out_size, void* d_ws, size_t ws_size,
                              hipStream_t stream) {
  (void)in_sizes; (void)n_in; (void)out_size; (void)ws_size;
  const float* img_e = (const float*)d_in[0];
  const float* txt_e = (const float*)d_in[1];
  const float* Wi    = (const float*)d_in[2];
  const float* bi    = (const float*)d_in[3];
  const float* Wt    = (const float*)d_in[4];
  const float* bt    = (const float*)d_in[5];
  const float* Wp    = (const float*)d_in[6];
  const float* bp    = (const float*)d_in[7];
  const float* Wc    = (const float*)d_in[8];
  const float* bc    = (const float*)d_in[9];

  float* ws     = (float*)d_ws;
  float* imgT   = ws;                         // 512*256 floats (transposed)
  float* txtN   = ws + 512 * 256;             // 256*512 floats
  float* pfeats = ws + 2 * 512 * 256;         // SEGS*256*512 floats (~8MB)

  map_norm_kernel<<<dim3(16, 2), 256, 0, stream>>>(img_e, txt_e, Wi, bi, Wt, bt,
                                                   imgT, txtN);
  bilinear_kernel<<<dim3(16 * SEGS), 256, 0, stream>>>(Wp, imgT, txtN, pfeats);
  head_kernel<<<dim3(256), 512, 0, stream>>>(pfeats, bp, Wc, bc, (float*)d_out);
}
